// PointsFusion_17755394801892
// MI455X (gfx1250) — compile-verified, matrix operand order checked by reading, not run on hardware
//
#include <hip/hip_runtime.h>
#include <hip/hip_bf16.h>
#include <math.h>

typedef __attribute__((ext_vector_type(16))) _Float16 v16h;
typedef __attribute__((ext_vector_type(8)))  float    v8f;

#define B_  4
#define N_  4096
#define K_  16          // k neighbors per ref cloud; 2k = 32 columns per point
#define S_  (B_ * N_ * 2 * K_)      // 524288 spatial columns
#define NTILES (S_ / 16)            // 32768 16-column WMMA tiles (2 per point)

__device__ __forceinline__ v8f wmma_f16(v16h a, v16h b, v8f c) {
    // D(16x16 f32) = A(16x32 f16) * B(32x16 f16) + C
    return __builtin_amdgcn_wmma_f32_16x16x32_f16(
        /*neg_a=*/false, a, /*neg_b=*/false, b,
        /*c_mod=*/(short)0, c, /*reuse_a=*/false, /*reuse_b=*/false);
}

// ---------------------------------------------------------------------------
// Fold BN into per-channel bias:  b' = scale*(b - mean) + beta, scale = g/sqrt(v+eps)
// biasAll layout: [0..63]=L0, [64..127]=L1, [128..255]=L2
// ---------------------------------------------------------------------------
__global__ __launch_bounds__(256) void bias_prep(
    const float* b0, const float* g0, const float* be0, const float* m0, const float* v0,
    const float* b1, const float* g1, const float* be1, const float* m1, const float* v1,
    const float* b2, const float* g2, const float* be2, const float* m2, const float* v2,
    float* biasAll) {
    int t = threadIdx.x;
    float bb, gg, be, mm, vv;
    if (t < 64)       { bb = b0[t];      gg = g0[t];      be = be0[t];      mm = m0[t];      vv = v0[t]; }
    else if (t < 128) { int i = t - 64;  bb = b1[i]; gg = g1[i]; be = be1[i]; mm = m1[i]; vv = v1[i]; }
    else              { int i = t - 128; bb = b2[i]; gg = g2[i]; be = be2[i]; mm = m2[i]; vv = v2[i]; }
    float sc = gg * rsqrtf(vv + 1e-3f);
    biasAll[t] = sc * (bb - mm) + be;
}

// ---------------------------------------------------------------------------
// Pack BN-folded weights into WMMA A-operand lane layout (f16).
// A-operand (16x32, f16): M = lane&15.  Half-slot s (0..15) of lane:
//   K = kt*32 + ((s<8)? s : s+8) + (lane>=16 ? 8 : 0)
// Tile table: tt 0..3 = L0 (mt=tt,kt=0); 4..11 = L1; 12..27 = L2.
// Each tile = 32 lanes x 16 halves = 1 KB, stored contiguously.
// ---------------------------------------------------------------------------
__global__ __launch_bounds__(32) void weight_prep(
    const float* w0, const float* g0, const float* v0,
    const float* w1, const float* g1, const float* v1,
    const float* w2, const float* g2, const float* v2,
    _Float16* Wa) {
    int tt   = blockIdx.x;      // 0..27
    int lane = threadIdx.x;     // 0..31
    const float *W, *G, *V;
    int Cin, mt, kt;
    if (tt < 4)       { W = w0; G = g0; V = v0; Cin = 4;  mt = tt;            kt = 0; }
    else if (tt < 12) { W = w1; G = g1; V = v1; Cin = 64; mt = (tt - 4) >> 1;  kt = (tt - 4) & 1; }
    else              { W = w2; G = g2; V = v2; Cin = 64; mt = (tt - 12) >> 1; kt = (tt - 12) & 1; }
    int m      = mt * 16 + (lane & 15);
    int laneHi = lane >> 4;
    float sc = G[m] * rsqrtf(V[m] + 1e-3f);
    v16h out = {};
#pragma unroll
    for (int s = 0; s < 16; ++s) {
        int k = kt * 32 + ((s < 8) ? s : s + 8) + laneHi * 8;
        float val = (k < Cin) ? sc * W[m * Cin + k] : 0.0f;
        out[s] = (_Float16)val;
    }
    ((v16h*)Wa)[tt * 32 + lane] = out;
}

// ---------------------------------------------------------------------------
// Brute-force KNN: one thread per (batch, refsel, query point).
// Ref points staged through LDS in 128-point chunks; register insertion sort
// keeps top-16 ascending (stable on ties, matching top_k tie-break).
// idx layout: [b][n][32] ints, j<16 from p1-refs, j>=16 from p2-refs.
// ---------------------------------------------------------------------------
__global__ __launch_bounds__(128) void knn_kernel(const float* __restrict__ p1,
                                                  const float* __restrict__ p2,
                                                  int* __restrict__ idx) {
    __shared__ float sx[128], sy[128], sz[128];
    int tid    = threadIdx.x;
    int blk    = blockIdx.x;          // 256 blocks: b(4) x refsel(2) x nchunk(32)
    int b      = blk >> 6;
    int rem    = blk & 63;
    int refsel = rem >> 5;
    int nchunk = rem & 31;
    int n      = nchunk * 128 + tid;

    const float* r = refsel ? p2 : p1;
    float qx = p1[(b * 3 + 0) * N_ + n];
    float qy = p1[(b * 3 + 1) * N_ + n];
    float qz = p1[(b * 3 + 2) * N_ + n];

    float best[16];
    int   bidx[16];
#pragma unroll
    for (int t = 0; t < 16; ++t) { best[t] = 3.0e38f; bidx[t] = 0; }

    for (int c0 = 0; c0 < N_; c0 += 128) {
        __syncthreads();
        int m = c0 + tid;
        sx[tid] = r[(b * 3 + 0) * N_ + m];
        sy[tid] = r[(b * 3 + 1) * N_ + m];
        sz[tid] = r[(b * 3 + 2) * N_ + m];
        __syncthreads();
        for (int i = 0; i < 128; ++i) {
            float dx = sx[i] - qx, dy = sy[i] - qy, dz = sz[i] - qz;
            float d2 = fmaf(dx, dx, fmaf(dy, dy, dz * dz));
            if (d2 < best[15]) {
                best[15] = d2; bidx[15] = c0 + i;
#pragma unroll
                for (int t = 15; t > 0; --t) {
                    if (best[t] < best[t - 1]) {
                        float td = best[t]; best[t] = best[t - 1]; best[t - 1] = td;
                        int   ti = bidx[t]; bidx[t] = bidx[t - 1]; bidx[t - 1] = ti;
                    }
                }
            }
        }
    }
    int base = (b * N_ + n) * 32 + refsel * 16;
#pragma unroll
    for (int t = 0; t < 16; ++t) idx[base + t] = bidx[t];
}

// ---------------------------------------------------------------------------
// Fused 3-layer MLP per wave (16 columns of one point's neighbor set).
// Features built in-register (only K rows 0..3 nonzero), layers chained via
// LDS-staged f16 B-operand repacks; channel max reduced in registers.
// ---------------------------------------------------------------------------
__global__ __launch_bounds__(256) void mlp_kernel(const float* __restrict__ p1,
                                                  const float* __restrict__ p2,
                                                  const int*   __restrict__ idx,
                                                  const _Float16* __restrict__ Wa,
                                                  const float* __restrict__ biasAll,
                                                  float* __restrict__ maxcol) {
    // per-wave double-buffered B-operand staging: [wave][stagebuf][kTile][lane][half]
    __shared__ __attribute__((aligned(32))) _Float16 stage[8][2][2][32][16];
    int lane = threadIdx.x & 31;
    int wave = threadIdx.x >> 5;
    int tile = blockIdx.x * 8 + wave;   // 0..32767
    int p    = tile >> 1;               // point index (b,n)
    int b    = p >> 12;
    int n    = p & (N_ - 1);
    int jBase = (tile & 1) << 4;
    const v16h* WaV = (const v16h*)Wa;

    // ---- Layer-0 B operand: features (resi.x,y,z, dist) in K rows 0..3 ----
    v16h fB = {};
    if (lane < 16) {
        int j  = jBase + lane;
        float qx = p1[(b * 3 + 0) * N_ + n];
        float qy = p1[(b * 3 + 1) * N_ + n];
        float qz = p1[(b * 3 + 2) * N_ + n];
        int   id = idx[p * 32 + j];
        const float* ref = (j < K_) ? p1 : p2;
        float dx = ref[(b * 3 + 0) * N_ + id] - qx;
        float dy = ref[(b * 3 + 1) * N_ + id] - qy;
        float dz = ref[(b * 3 + 2) * N_ + id] - qz;
        float d2 = fmaf(dx, dx, fmaf(dy, dy, dz * dz));
        float dist = sqrtf(fmaxf(d2, 1e-12f));
        fB[0] = (_Float16)dx; fB[1] = (_Float16)dy;
        fB[2] = (_Float16)dz; fB[3] = (_Float16)dist;
    }

    // ---- Layer 0: 4->64 (K padded to 32), 4 M-tiles ----
#pragma unroll
    for (int mt = 0; mt < 4; ++mt) {
        v16h a   = WaV[(0 + mt) * 32 + lane];
        v8f  acc = {};
        acc = wmma_f16(a, fB, acc);
#pragma unroll
        for (int r = 0; r < 8; ++r) {
            int mloc = (lane < 16) ? r : r + 8;
            int m    = mt * 16 + mloc;
            int nn   = lane & 15;
            float y  = fmaxf(acc[r] + biasAll[m], 0.0f);
            int kk = m & 31, kt = m >> 5;
            int cl   = (((kk >> 3) & 1) ? 16 : 0) + nn;
            int slot = (kk & 7) | ((kk >> 1) & 8);
            stage[wave][0][kt][cl][slot] = (_Float16)y;
        }
    }
    __builtin_amdgcn_wave_barrier();
    asm volatile("s_wait_dscnt 0" ::: "memory");
    v16h b1lo = *(const v16h*)&stage[wave][0][0][lane][0];
    v16h b1hi = *(const v16h*)&stage[wave][0][1][lane][0];

    // ---- Layer 1: 64->64, K=64 = 2 chained WMMAs per M-tile ----
#pragma unroll
    for (int mt = 0; mt < 4; ++mt) {
        v16h a0  = WaV[(4 + mt * 2 + 0) * 32 + lane];
        v16h a1  = WaV[(4 + mt * 2 + 1) * 32 + lane];
        v8f  acc = {};
        acc = wmma_f16(a0, b1lo, acc);
        acc = wmma_f16(a1, b1hi, acc);
#pragma unroll
        for (int r = 0; r < 8; ++r) {
            int mloc = (lane < 16) ? r : r + 8;
            int m    = mt * 16 + mloc;
            int nn   = lane & 15;
            float y  = fmaxf(acc[r] + biasAll[64 + m], 0.0f);
            int kk = m & 31, kt = m >> 5;
            int cl   = (((kk >> 3) & 1) ? 16 : 0) + nn;
            int slot = (kk & 7) | ((kk >> 1) & 8);
            stage[wave][1][kt][cl][slot] = (_Float16)y;
        }
    }
    __builtin_amdgcn_wave_barrier();
    asm volatile("s_wait_dscnt 0" ::: "memory");
    v16h b2lo = *(const v16h*)&stage[wave][1][0][lane][0];
    v16h b2hi = *(const v16h*)&stage[wave][1][1][lane][0];

    // ---- Layer 2: 64->128, 8 M-tiles; fuse channel-max into epilogue ----
    float mval = -3.0e38f;
#pragma unroll
    for (int mt = 0; mt < 8; ++mt) {
        v16h a0  = WaV[(12 + mt * 2 + 0) * 32 + lane];
        v16h a1  = WaV[(12 + mt * 2 + 1) * 32 + lane];
        v8f  acc = {};
        acc = wmma_f16(a0, b2lo, acc);
        acc = wmma_f16(a1, b2hi, acc);
#pragma unroll
        for (int r = 0; r < 8; ++r) {
            int mloc = (lane < 16) ? r : r + 8;
            float y  = fmaxf(acc[r] + biasAll[128 + mt * 16 + mloc], 0.0f);
            mval = fmaxf(mval, y);
        }
    }
    // column n's channels live in lanes n and n+16: combine and store
    mval = fmaxf(mval, __shfl_xor(mval, 16, 32));
    if (lane < 16) maxcol[tile * 16 + lane] = mval;
}

// ---------------------------------------------------------------------------
// Per point: softmax over 32 channel-max scores, weighted sum of gathered
// neighbor coordinates -> out (B,3,N). One wave per point.
// ---------------------------------------------------------------------------
__global__ __launch_bounds__(256) void fuse_kernel(const float* __restrict__ p1,
                                                   const float* __restrict__ p2,
                                                   const int*   __restrict__ idx,
                                                   const float* __restrict__ maxcol,
                                                   float* __restrict__ out) {
    int lane = threadIdx.x & 31;
    int wv   = threadIdx.x >> 5;
    int p    = blockIdx.x * 8 + wv;     // 0..16383
    int b    = p >> 12;
    int n    = p & (N_ - 1);

    float x  = maxcol[p * 32 + lane];
    float mx = x;
#pragma unroll
    for (int o = 16; o; o >>= 1) mx = fmaxf(mx, __shfl_xor(mx, o, 32));
    float e = __expf(x - mx);
    float s = e;
#pragma unroll
    for (int o = 16; o; o >>= 1) s += __shfl_xor(s, o, 32);
    float w = e / s;

    int id = idx[p * 32 + lane];
    const float* ref = (lane < K_) ? p1 : p2;
#pragma unroll
    for (int c = 0; c < 3; ++c) {
        float v = w * ref[(b * 3 + c) * N_ + id];
#pragma unroll
        for (int o = 16; o; o >>= 1) v += __shfl_xor(v, o, 32);
        if (lane == 0) out[(b * 3 + c) * N_ + n] = v;
    }
}

// ---------------------------------------------------------------------------
extern "C" void kernel_launch(void* const* d_in, const int* in_sizes, int n_in,
                              void* d_out, int out_size, void* d_ws, size_t ws_size,
                              hipStream_t stream) {
    const float* p1 = (const float*)d_in[0];
    const float* p2 = (const float*)d_in[1];
    // d_in[2] = k (scalar, fixed 16 per setup)
    const float* w0 = (const float*)d_in[3];
    const float* b0 = (const float*)d_in[4];
    const float* g0 = (const float*)d_in[5];
    const float* be0 = (const float*)d_in[6];
    const float* m0 = (const float*)d_in[7];
    const float* v0 = (const float*)d_in[8];
    const float* w1 = (const float*)d_in[9];
    const float* b1 = (const float*)d_in[10];
    const float* g1 = (const float*)d_in[11];
    const float* be1 = (const float*)d_in[12];
    const float* m1 = (const float*)d_in[13];
    const float* v1 = (const float*)d_in[14];
    const float* w2 = (const float*)d_in[15];
    const float* b2 = (const float*)d_in[16];
    const float* g2 = (const float*)d_in[17];
    const float* be2 = (const float*)d_in[18];
    const float* m2 = (const float*)d_in[19];
    const float* v2 = (const float*)d_in[20];
    float* outp = (float*)d_out;

    char* ws = (char*)d_ws;
    int*       idx     = (int*)ws;                                  // 2 MB
    float*     maxcol  = (float*)(ws + (size_t)(2 << 20));          // 2 MB
    _Float16*  Wa      = (_Float16*)(ws + (size_t)(4 << 20));       // 28 KB packed
    float*     biasAll = (float*)(ws + (size_t)(4 << 20) + 32768);  // 1 KB

    bias_prep<<<1, 256, 0, stream>>>(b0, g0, be0, m0, v0,
                                     b1, g1, be1, m1, v1,
                                     b2, g2, be2, m2, v2, biasAll);
    weight_prep<<<28, 32, 0, stream>>>(w0, g0, v0, w1, g1, v1, w2, g2, v2, Wa);
    knn_kernel<<<256, 128, 0, stream>>>(p1, p2, idx);
    mlp_kernel<<<NTILES / 8, 256, 0, stream>>>(p1, p2, idx, Wa, biasAll, maxcol);
    fuse_kernel<<<(B_ * N_) / 8, 256, 0, stream>>>(p1, p2, idx, maxcol, outp);
}
